// Transformer_34419867910340
// MI455X (gfx1250) — matrix operand order repaired and checked by reading, not compiled, not run on hardware
//
#include <hip/hip_runtime.h>
#include <math.h>

// ---------------------------------------------------------------------------
// Transformer forward (4 layers) for gfx1250 / MI455X.
// bf16 WMMA GEMMs, TDM (tensor_load_to_lds) staged A-operand, LDS attention.
// ---------------------------------------------------------------------------

typedef unsigned short u16;
typedef unsigned int   u32;
typedef __attribute__((ext_vector_type(16))) __bf16 v16bf;
typedef __attribute__((ext_vector_type(8)))  float  v8f;
typedef __attribute__((ext_vector_type(8)))  int    v8i;
typedef __attribute__((ext_vector_type(4)))  int    v4i;
typedef __attribute__((ext_vector_type(4)))  unsigned int u32x4;
typedef __attribute__((ext_vector_type(8)))  int    i32x8;
typedef __attribute__((ext_vector_type(4)))  int    i32x4;

#define DIM    1024
#define HEADS  16
#define DHEAD  64
#define INNER  1024
#define FF     4096
#define SEQ    1024
#define BATCH  2
#define ROWS   (BATCH * SEQ)          // 2048
#define QK_SCALE 0.125f               // 64^-0.5

// ---- helpers ---------------------------------------------------------------

static __device__ __forceinline__ u16 f2bf(float f) {
  u32 u = __builtin_bit_cast(u32, f);
  u32 r = (u + 0x7FFFu + ((u >> 16) & 1u)) >> 16;   // round-to-nearest-even
  return (u16)r;
}

// A-fragment: 16x32 bf16, row-major source with leading dim `ld` (elements).
static __device__ __forceinline__ v16bf load_frag_a(const u16* base, int ld, int lane) {
  const int m  = lane & 15;
  const int hf = lane >> 4;
  const u16* p = base + (size_t)m * ld + hf * 8;
  v4i lo = *(const v4i*)p;          // K = c0 .. c0+7
  v4i hi = *(const v4i*)(p + 16);   // K = c0+16 .. c0+23
  v8i t;
  t[0] = lo[0]; t[1] = lo[1]; t[2] = lo[2]; t[3] = lo[3];
  t[4] = hi[0]; t[5] = hi[1]; t[6] = hi[2]; t[7] = hi[3];
  return __builtin_bit_cast(v16bf, t);
}

// B-fragment: 32x16 bf16 from B^T storage [N][K] with leading dim `ld`.
static __device__ __forceinline__ v16bf load_frag_b(const u16* base, int ld, int lane) {
  const int n  = lane & 15;
  const int hf = lane >> 4;
  const u16* p = base + (size_t)n * ld + hf * 8;
  v4i lo = *(const v4i*)p;
  v4i hi = *(const v4i*)(p + 16);
  v8i t;
  t[0] = lo[0]; t[1] = lo[1]; t[2] = lo[2]; t[3] = lo[3];
  t[4] = hi[0]; t[5] = hi[1]; t[6] = hi[2]; t[7] = hi[3];
  return __builtin_bit_cast(v16bf, t);
}

static __device__ __forceinline__ v8f wmma_bf16(v16bf a, v16bf b, v8f c) {
  return __builtin_amdgcn_wmma_f32_16x16x32_bf16(
      false, a, false, b, (short)0, c, false, false);
}

// ---- TDM: DMA a 2D bf16 tile [tile_rows x 128] (row stride = stride_elems)
//      from global into LDS at byte offset lds_off. D# per CDNA5 ISA 8.3/8.4.
static __device__ __forceinline__ void tdm_load_2d(
    unsigned lds_off, const void* gptr, int tile_rows, int stride_elems) {
  unsigned long long ga = (unsigned long long)(size_t)gptr;
  u32x4 g0;
  g0[0] = 1u;                                            // count=1
  g0[1] = lds_off;                                       // lds_addr
  g0[2] = (unsigned)ga;                                  // global_addr[31:0]
  g0[3] = ((unsigned)(ga >> 32) & 0x01FFFFFFu)           // global_addr[56:32]
        | 0x80000000u;                                   // type=2 (image)
  i32x8 g1;
  g1[0] = (int)(1u << 16);          // data_size=1 (2 bytes), mask=0, no pad
  g1[1] = 0;                        // tensor_dim0 low16 = 0 (dim0 = 0x40000000)
  g1[2] = (int)(0x4000u);           // tensor_dim0 hi16; tensor_dim1 low16 = 0
  g1[3] = (int)(0x10u | (128u << 16)); // tensor_dim1 hi16 (=1M); tile_dim0=128
  g1[4] = tile_rows;                // tile_dim1; tile_dim2=0
  g1[5] = stride_elems;             // tensor_dim0_stride low32
  g1[6] = 0;                        // stride hi16; tensor_dim1_stride low16
  g1[7] = 0;                        // tensor_dim1_stride hi32
  i32x4 z4 = {0, 0, 0, 0};
#if defined(__clang_major__) && (__clang_major__ >= 23)
  i32x8 z8 = {0, 0, 0, 0, 0, 0, 0, 0};
  __builtin_amdgcn_tensor_load_to_lds(g0, g1, z4, z4, z8, 0);
#else
  __builtin_amdgcn_tensor_load_to_lds(g0, g1, z4, z4, 0);
#endif
}

// ---- LayerNorm -> bf16 -----------------------------------------------------

__global__ __launch_bounds__(256) void layernorm_to_bf16(
    const float* __restrict__ x, const float* __restrict__ w,
    const float* __restrict__ b, u16* __restrict__ out) {
  __shared__ float red[16];
  const int row = blockIdx.x;
  const int tid = threadIdx.x;
  const float* xr = x + (size_t)row * DIM;
  float s = 0.f, s2 = 0.f;
#pragma unroll
  for (int k = 0; k < 4; ++k) {
    float v = xr[tid + k * 256];
    s += v; s2 += v * v;
  }
#pragma unroll
  for (int o = 16; o >= 1; o >>= 1) {
    s  += __shfl_xor(s,  o, 32);
    s2 += __shfl_xor(s2, o, 32);
  }
  if ((tid & 31) == 0) { red[(tid >> 5) * 2] = s; red[(tid >> 5) * 2 + 1] = s2; }
  __syncthreads();
  if (tid == 0) {
    float ts = 0.f, t2 = 0.f;
    for (int wv = 0; wv < 8; ++wv) { ts += red[wv * 2]; t2 += red[wv * 2 + 1]; }
    red[0] = ts; red[1] = t2;
  }
  __syncthreads();
  const float mu   = red[0] * (1.0f / DIM);
  const float var  = red[1] * (1.0f / DIM) - mu * mu;
  const float rstd = rsqrtf(var + 1e-5f);
#pragma unroll
  for (int k = 0; k < 4; ++k) {
    int j = tid + k * 256;
    out[(size_t)row * DIM + j] = f2bf((xr[j] - mu) * rstd * w[j] + b[j]);
  }
}

// ---- f32 [K,N] weight -> bf16 [N,K] transpose (K = 1<<kshift) --------------

__global__ __launch_bounds__(256) void transpose_cvt(
    const float* __restrict__ W, u16* __restrict__ WT, int kshift, int N) {
  size_t idx = (size_t)blockIdx.x * 256 + threadIdx.x;   // idx = n*K + k
  int k = (int)(idx & (((size_t)1 << kshift) - 1));
  int n = (int)(idx >> kshift);
  WT[idx] = f2bf(W[(size_t)k * N + n]);
}

// ---- GEMM: C[M,N](f32) = A_bf16[M,K] @ (BT_bf16[N,K])^T  (+ optional bias) -
// 256 threads = 8 waves, block tile 128(M) x 64(N), wave tile 16x64.
// A is staged through LDS in 128x128 chunks by the Tensor Data Mover,
// double-buffered (2 x 32KB), synchronized with s_wait_tensorcnt + barrier.
// B fragments are register double-buffered (prefetch k+32 during WMMA at k).

__global__ __launch_bounds__(256) void gemm_bf16_wmma(
    const u16* __restrict__ A, const u16* __restrict__ BT,
    float* __restrict__ C, const float* __restrict__ bias, int N, int K) {
  extern __shared__ char gemm_lds[];
  u16* ldsA = (u16*)gemm_lds;               // [2][128][128] bf16

  const int tid  = threadIdx.x;
  const int wave = tid >> 5;
  const int lane = tid & 31;
  const int m0   = blockIdx.y * 128;
  const int n0   = blockIdx.x * 64;

  // chunk 0 of A -> LDS buffer 0 via TDM
  if (wave == 0) {
    tdm_load_2d(0u, A + (size_t)m0 * K, 128, K);
    __builtin_amdgcn_s_wait_tensorcnt(0);
  }
  __syncthreads();

  v8f acc[4] = {};
  // B pipeline: preload k = 0 fragments
  v16bf bc[4];
#pragma unroll
  for (int t = 0; t < 4; ++t)
    bc[t] = load_frag_b(BT + (size_t)(n0 + t * 16) * K, K, lane);

  int buf = 0;
  for (int kc = 0; kc < K; kc += 128) {
    const bool more = (kc + 128) < K;
    if (wave == 0 && more)
      tdm_load_2d((unsigned)((buf ^ 1) * 32768), A + (size_t)m0 * K + kc + 128,
                  128, K);
    const u16* Abuf = ldsA + buf * 16384 + (size_t)(wave * 16) * 128;
#pragma unroll
    for (int ks = 0; ks < 4; ++ks) {
      int knext = kc + ks * 32 + 32;
      if (knext >= K) knext = 0;                 // dummy (value unused)
      v16bf bn[4];
#pragma unroll
      for (int t = 0; t < 4; ++t)
        bn[t] = load_frag_b(BT + (size_t)(n0 + t * 16) * K + knext, K, lane);
      v16bf a = load_frag_a(Abuf + ks * 32, 128, lane);
#pragma unroll
      for (int t = 0; t < 4; ++t) acc[t] = wmma_bf16(a, bc[t], acc[t]);
#pragma unroll
      for (int t = 0; t < 4; ++t) bc[t] = bn[t];
    }
    if (wave == 0 && more) __builtin_amdgcn_s_wait_tensorcnt(0);
    __syncthreads();
    buf ^= 1;
  }

  const int hf = lane >> 4, nn = lane & 15;
#pragma unroll
  for (int t = 0; t < 4; ++t) {
    int col = n0 + t * 16 + nn;
    float bv = bias ? bias[col] : 0.0f;
#pragma unroll
    for (int r = 0; r < 8; ++r) {
      C[(size_t)(m0 + wave * 16 + hf * 8 + r) * N + col] = acc[t][r] + bv;
    }
  }
}

// ---- split qkv: f32 [ROWS,3*INNER] -> qb/kb [bh][n][64] bf16 (q*scale),
//      v transposed to vt [bh][64][n] bf16 ----------------------------------

__global__ __launch_bounds__(256) void split_qkv(
    const float* __restrict__ qkv, u16* __restrict__ qb,
    u16* __restrict__ kb, u16* __restrict__ vt) {
  int idx = blockIdx.x * 256 + threadIdx.x;   // over ROWS*INNER
  int row = idx >> 10;
  int col = idx & 1023;
  int b  = row >> 10, nn = row & 1023;
  int h  = col >> 6,  d  = col & 63;
  int bh = b * HEADS + h;
  const float* p = qkv + (size_t)row * (3 * INNER);
  qb[((size_t)bh * SEQ + nn) * DHEAD + d] = f2bf(p[col] * QK_SCALE);
  kb[((size_t)bh * SEQ + nn) * DHEAD + d] = f2bf(p[INNER + col]);
  vt[((size_t)bh * DHEAD + d) * SEQ + nn] = f2bf(p[2 * INNER + col]);
}

// ---- attention: per (b,h, 16-query block). S in LDS, softmax, O = P @ V ----
// dynamic LDS: S f32 [16][1024] (64KB) + P bf16 [16][1024] (32KB) = 96KB.

__global__ __launch_bounds__(256) void attn_kernel(
    const u16* __restrict__ qb, const u16* __restrict__ kb,
    const u16* __restrict__ vt, u16* __restrict__ ab) {
  extern __shared__ char attn_lds[];
  float* S = (float*)attn_lds;                   // [16][SEQ]
  u16*   P = (u16*)(attn_lds + 16 * SEQ * 4);    // [16][SEQ]

  const int bh = blockIdx.y;                     // b*16 + h
  const int b  = bh >> 4, h = bh & 15;
  const int q0 = blockIdx.x * 16;
  const int tid  = threadIdx.x;
  const int wave = tid >> 5;
  const int lane = tid & 31;

  const u16* Q  = qb + (size_t)bh * SEQ * DHEAD;
  const u16* Kb = kb + (size_t)bh * SEQ * DHEAD;
  const u16* Vt = vt + (size_t)bh * DHEAD * SEQ;

  // phase 1: S[16][SEQ] = Qblk @ K^T  (each wave: 16-col tiles, stride 128)
  v16bf a0 = load_frag_a(Q + (size_t)q0 * DHEAD + 0,  DHEAD, lane);
  v16bf a1 = load_frag_a(Q + (size_t)q0 * DHEAD + 32, DHEAD, lane);
  const int hf = lane >> 4, nn = lane & 15;
  for (int jt = wave * 16; jt < SEQ; jt += 128) {
    v16bf b0 = load_frag_b(Kb + (size_t)jt * DHEAD + 0,  DHEAD, lane);
    v16bf b1 = load_frag_b(Kb + (size_t)jt * DHEAD + 32, DHEAD, lane);
    v8f acc = {};
    acc = wmma_bf16(a0, b0, acc);
    acc = wmma_bf16(a1, b1, acc);
#pragma unroll
    for (int r = 0; r < 8; ++r) S[(hf * 8 + r) * SEQ + jt + nn] = acc[r];
  }
  __syncthreads();

  // phase 2: softmax, 16 threads per row, 64 cols each; reduce via shfl(w=16)
  {
    const int r = tid >> 4, c = tid & 15;
    float* Srow = S + r * SEQ + c * 64;
    float mx = -1e30f;
    for (int j = 0; j < 64; ++j) mx = fmaxf(mx, Srow[j]);
#pragma unroll
    for (int o = 8; o >= 1; o >>= 1) mx = fmaxf(mx, __shfl_xor(mx, o, 16));
    float sum = 0.f;
    for (int j = 0; j < 64; ++j) { float e = __expf(Srow[j] - mx); Srow[j] = e; sum += e; }
#pragma unroll
    for (int o = 8; o >= 1; o >>= 1) sum += __shfl_xor(sum, o, 16);
    const float inv = 1.0f / sum;
    u16* Prow = P + r * SEQ + c * 64;
    for (int j = 0; j < 64; ++j) Prow[j] = f2bf(Srow[j] * inv);
  }
  __syncthreads();

  // phase 3: O[16][64] = P @ V  (waves 0..3, each a 16-wide d tile)
  if (wave < 4) {
    const int wn = wave;
    v8f acc = {};
    for (int k0 = 0; k0 < SEQ; k0 += 32) {
      v16bf a  = load_frag_a(P + k0, SEQ, lane);
      v16bf bb = load_frag_b(Vt + (size_t)(wn * 16) * SEQ + k0, SEQ, lane);
      acc = wmma_bf16(a, bb, acc);
    }
#pragma unroll
    for (int r = 0; r < 8; ++r) {
      int row = b * SEQ + q0 + hf * 8 + r;
      int col = h * DHEAD + wn * 16 + nn;
      ab[(size_t)row * INNER + col] = f2bf(acc[r]);
    }
  }
}

// ---- GEGLU: u f32 [ROWS, 2*FF] (+bias) -> a*gelu(g) bf16 [ROWS, FF] --------

__global__ __launch_bounds__(256) void geglu_kernel(
    const float* __restrict__ u, const float* __restrict__ bff1,
    u16* __restrict__ out) {
  size_t idx = (size_t)blockIdx.x * 256 + threadIdx.x;  // over ROWS*FF
  int row = (int)(idx >> 12);
  int col = (int)(idx & 4095);
  const float* ur = u + (size_t)row * (2 * FF);
  float a = ur[col] + bff1[col];
  float g = ur[FF + col] + bff1[FF + col];
  float gelu = 0.5f * g * (1.0f + erff(g * 0.70710678118654752f));
  out[idx] = f2bf(a * gelu);
}

// ---- residual + LayerScale: x += o * ls ------------------------------------

__global__ __launch_bounds__(256) void residual_ls(
    float* __restrict__ x, const float* __restrict__ o,
    const float* __restrict__ ls) {
  size_t idx = (size_t)blockIdx.x * 256 + threadIdx.x;  // over ROWS*DIM
  x[idx] += o[idx] * ls[idx & (DIM - 1)];
}

// ---------------------------------------------------------------------------

extern "C" void kernel_launch(void* const* d_in, const int* in_sizes, int n_in,
                              void* d_out, int out_size, void* d_ws, size_t ws_size,
                              hipStream_t stream) {
  (void)in_sizes; (void)n_in; (void)out_size; (void)ws_size;

  const float* x_in = (const float*)d_in[0];
  const float* LN1W = (const float*)d_in[1];
  const float* LN1B = (const float*)d_in[2];
  const float* WQKV = (const float*)d_in[3];
  const float* WOUT = (const float*)d_in[4];
  const float* BOUT = (const float*)d_in[5];
  const float* LS1  = (const float*)d_in[6];
  const float* LN2W = (const float*)d_in[7];
  const float* LN2B = (const float*)d_in[8];
  const float* WFF1 = (const float*)d_in[9];
  const float* BFF1 = (const float*)d_in[10];
  const float* WFF2 = (const float*)d_in[11];
  const float* BFF2 = (const float*)d_in[12];
  const float* LS2  = (const float*)d_in[13];

  // workspace layout
  char* wp = (char*)d_ws;
  auto alloc = [&](size_t bytes) -> void* {
    void* p = wp;
    wp += (bytes + 255) & ~(size_t)255;
    return p;
  };
  float* xcur = (float*)alloc((size_t)ROWS * DIM * 4);
  u16*   hb   = (u16*)  alloc((size_t)ROWS * DIM * 2);
  u16*   wT   = (u16*)  alloc((size_t)8192 * 1024 * 2);     // largest weight
  float* qkvf = (float*)alloc((size_t)ROWS * 3 * INNER * 4);
  u16*   qb   = (u16*)  alloc((size_t)BATCH * HEADS * SEQ * DHEAD * 2);
  u16*   kbuf = (u16*)  alloc((size_t)BATCH * HEADS * SEQ * DHEAD * 2);
  u16*   vtb  = (u16*)  alloc((size_t)BATCH * HEADS * DHEAD * SEQ * 2);
  u16*   ab   = (u16*)  alloc((size_t)ROWS * INNER * 2);
  float* of   = (float*)alloc((size_t)ROWS * DIM * 4);
  float* ubuf = (float*)alloc((size_t)ROWS * 2 * FF * 4);
  u16*   gg   = (u16*)  alloc((size_t)ROWS * FF * 2);
  float* ff   = (float*)alloc((size_t)ROWS * DIM * 4);

  hipMemcpyAsync(xcur, x_in, (size_t)ROWS * DIM * 4,
                 hipMemcpyDeviceToDevice, stream);

  const dim3 blk(256);
  const size_t gemm_smem = 2 * 128 * 128 * 2;                            // 64KB
  const size_t attn_smem = (size_t)16 * SEQ * 4 + (size_t)16 * SEQ * 2;  // 96KB

  for (int l = 0; l < 4; ++l) {
    const float* ln1w = LN1W + (size_t)l * DIM;
    const float* ln1b = LN1B + (size_t)l * DIM;
    const float* wqkv = WQKV + (size_t)l * DIM * 3 * INNER;
    const float* wout = WOUT + (size_t)l * INNER * DIM;
    const float* bout = BOUT + (size_t)l * DIM;
    const float* ls1  = LS1  + (size_t)l * DIM;
    const float* ln2w = LN2W + (size_t)l * DIM;
    const float* ln2b = LN2B + (size_t)l * DIM;
    const float* wff1 = WFF1 + (size_t)l * DIM * 2 * FF;
    const float* bff1 = BFF1 + (size_t)l * 2 * FF;
    const float* wff2 = WFF2 + (size_t)l * FF * DIM;
    const float* bff2 = BFF2 + (size_t)l * DIM;
    const float* ls2  = LS2  + (size_t)l * DIM;

    // --- attention branch ---
    layernorm_to_bf16<<<ROWS, blk, 0, stream>>>(xcur, ln1w, ln1b, hb);
    transpose_cvt<<<(DIM * 3 * INNER) / 256, blk, 0, stream>>>(wqkv, wT, 10, 3 * INNER);
    gemm_bf16_wmma<<<dim3((3 * INNER) / 64, ROWS / 128), blk, gemm_smem, stream>>>(
        hb, wT, qkvf, nullptr, 3 * INNER, DIM);
    split_qkv<<<(ROWS * INNER) / 256, blk, 0, stream>>>(qkvf, qb, kbuf, vtb);
    attn_kernel<<<dim3(SEQ / 16, BATCH * HEADS), blk, attn_smem, stream>>>(
        qb, kbuf, vtb, ab);
    transpose_cvt<<<(INNER * DIM) / 256, blk, 0, stream>>>(wout, wT, 10, DIM);
    gemm_bf16_wmma<<<dim3(DIM / 64, ROWS / 128), blk, gemm_smem, stream>>>(
        ab, wT, of, bout, DIM, INNER);
    residual_ls<<<(ROWS * DIM) / 256, blk, 0, stream>>>(xcur, of, ls1);

    // --- feed-forward branch ---
    layernorm_to_bf16<<<ROWS, blk, 0, stream>>>(xcur, ln2w, ln2b, hb);
    transpose_cvt<<<(DIM * 2 * FF) / 256, blk, 0, stream>>>(wff1, wT, 10, 2 * FF);
    gemm_bf16_wmma<<<dim3((2 * FF) / 64, ROWS / 128), blk, gemm_smem, stream>>>(
        hb, wT, ubuf, nullptr, 2 * FF, DIM);
    geglu_kernel<<<(ROWS * FF) / 256, blk, 0, stream>>>(ubuf, bff1, gg);
    transpose_cvt<<<(FF * DIM) / 256, blk, 0, stream>>>(wff2, wT, 12, DIM);
    gemm_bf16_wmma<<<dim3(DIM / 64, ROWS / 128), blk, gemm_smem, stream>>>(
        gg, wT, ff, bff2, DIM, FF);
    residual_ls<<<(ROWS * DIM) / 256, blk, 0, stream>>>(xcur, ff, ls2);
  }

  hipMemcpyAsync(d_out, xcur, (size_t)ROWS * DIM * 4,
                 hipMemcpyDeviceToDevice, stream);
}
